// VanillaRNN_23063974380056
// MI455X (gfx1250) — matrix operand order repaired and checked by reading, not compiled, not run on hardware
//
#include <hip/hip_runtime.h>
#include <cmath>

// CDNA5 / gfx1250: wave32, WMMA 16x16x32 bf16 with f32 accumulation.
typedef __attribute__((ext_vector_type(16))) __bf16 v16bf;
typedef __attribute__((ext_vector_type(8)))  __bf16 v8bf;
typedef __attribute__((ext_vector_type(8)))  float  v8f;
typedef __attribute__((ext_vector_type(4)))  int    v4i;
typedef __attribute__((address_space(3)))    v4i    lds_v4i;   // LDS-qualified

#define WMMA_BF16(a, b, c) \
  __builtin_amdgcn_wmma_f32_16x16x32_bf16(false, (a), false, (b), (short)0, (c), false, false)

#if __has_builtin(__builtin_amdgcn_tanhf)
#define TANHF(x) __builtin_amdgcn_tanhf(x)   // v_tanh_f32 (trans op)
#else
#define TANHF(x) tanhf(x)
#endif

#if __has_builtin(__builtin_amdgcn_global_load_async_to_lds_b128)
#define HAVE_ASYNC_LDS 1
#else
#define HAVE_ASYNC_LDS 0
#endif

static __device__ __forceinline__ void wait_async_zero() {
#if __has_builtin(__builtin_amdgcn_s_wait_asynccnt)
  __builtin_amdgcn_s_wait_asynccnt(0);
#else
  asm volatile("s_wait_asynccnt 0" ::: "memory");
#endif
}

// ---------- fragment builders (per CDNA5 ISA 7.12.2 layouts) ----------------
// A (16x32, 16-bit): lanes 0-15 -> row=lane, K {0..7,16..23}; lanes 16-31 ->
// row=lane-16, K {8..15,24..31}.
static __device__ __forceinline__ v16bf fragA_f32g(const float* __restrict__ tile, int ld) {
  const int lane = threadIdx.x & 31;
  const float* p = tile + (size_t)(lane & 15) * ld + ((lane >> 4) << 3);
  const float4 a0 = *(const float4*)(p + 0);
  const float4 a1 = *(const float4*)(p + 4);
  const float4 a2 = *(const float4*)(p + 16);
  const float4 a3 = *(const float4*)(p + 20);
  v16bf f;
  f[ 0]=(__bf16)a0.x; f[ 1]=(__bf16)a0.y; f[ 2]=(__bf16)a0.z; f[ 3]=(__bf16)a0.w;
  f[ 4]=(__bf16)a1.x; f[ 5]=(__bf16)a1.y; f[ 6]=(__bf16)a1.z; f[ 7]=(__bf16)a1.w;
  f[ 8]=(__bf16)a2.x; f[ 9]=(__bf16)a2.y; f[10]=(__bf16)a2.z; f[11]=(__bf16)a2.w;
  f[12]=(__bf16)a3.x; f[13]=(__bf16)a3.y; f[14]=(__bf16)a3.z; f[15]=(__bf16)a3.w;
  return f;
}

// B (32x16, 16-bit): lanes 0-15 -> col=lane, K {0..7,16..23}; lanes 16-31 -> col=lane-16,
// K {8..15,24..31}.  Source rows of Wsrc are the output columns (B[k][j] = Wsrc[j][k]).
static __device__ __forceinline__ v16bf fragB_f32g(const float* __restrict__ Wsrc, int ld,
                                                   int j0, int k0) {
  const int lane = threadIdx.x & 31;
  const float* p = Wsrc + (size_t)(j0 + (lane & 15)) * ld + k0 + ((lane >> 4) << 3);
  const float4 b0 = *(const float4*)(p + 0);
  const float4 b1 = *(const float4*)(p + 4);
  const float4 b2 = *(const float4*)(p + 16);
  const float4 b3 = *(const float4*)(p + 20);
  v16bf f;
  f[ 0]=(__bf16)b0.x; f[ 1]=(__bf16)b0.y; f[ 2]=(__bf16)b0.z; f[ 3]=(__bf16)b0.w;
  f[ 4]=(__bf16)b1.x; f[ 5]=(__bf16)b1.y; f[ 6]=(__bf16)b1.z; f[ 7]=(__bf16)b1.w;
  f[ 8]=(__bf16)b2.x; f[ 9]=(__bf16)b2.y; f[10]=(__bf16)b2.z; f[11]=(__bf16)b2.w;
  f[12]=(__bf16)b3.x; f[13]=(__bf16)b3.y; f[14]=(__bf16)b3.z; f[15]=(__bf16)b3.w;
  return f;
}

// ---------------------------------------------------------------------------
// Kernel 1: x_proj[bt, m] = sum_n input[bt, n] * W_input[m, n], written into
// d_out in place.  HBM-bound (~132 MB @ 23.3 TB/s ~ 6 us); bf16 WMMA, f32
// accum, on-the-fly f32->bf16 (minimizes total bytes vs a pre-convert pass).
// ---------------------------------------------------------------------------
__global__ __launch_bounds__(256, 1) void xproj_kernel(const float* __restrict__ X,
                                                       const float* __restrict__ Wi,
                                                       float* __restrict__ out) {
  constexpr int N = 1024;
  const int bm = blockIdx.x & 127;       // 16384 / 128
  const int bn = blockIdx.x >> 7;        // 1024 / 64
  const int wave = threadIdx.x >> 5;
  const int m0 = bm * 128 + (wave & 3) * 32;
  const int n0 = bn * 64 + (wave >> 2) * 32;

  v8f c00 = {0,0,0,0,0,0,0,0}, c01 = c00, c10 = c00, c11 = c00;

  for (int k0 = 0; k0 < N; k0 += 32) {
    const v16bf a0 = fragA_f32g(X + (size_t)m0 * N + k0, N);
    const v16bf a1 = fragA_f32g(X + (size_t)(m0 + 16) * N + k0, N);
    const v16bf b0 = fragB_f32g(Wi, N, n0, k0);
    const v16bf b1 = fragB_f32g(Wi, N, n0 + 16, k0);
    c00 = WMMA_BF16(a0, b0, c00);
    c01 = WMMA_BF16(a0, b1, c01);
    c10 = WMMA_BF16(a1, b0, c10);
    c11 = WMMA_BF16(a1, b1, c11);
  }

  const int lane = threadIdx.x & 31;
  const int col = lane & 15;
  const int mb = (lane >> 4) << 3;
#pragma unroll
  for (int r = 0; r < 8; ++r) {
    const int m = mb + r;
    out[(size_t)(m0 + m) * N + n0 + col]           = c00[r];
    out[(size_t)(m0 + m) * N + n0 + 16 + col]      = c01[r];
    out[(size_t)(m0 + 16 + m) * N + n0 + col]      = c10[r];
    out[(size_t)(m0 + 16 + m) * N + n0 + 16 + col] = c11[r];
  }
}

// ---------------------------------------------------------------------------
// Kernel 2: sequential scan.  32 persistent WGs (2 batch-tiles x 16 column
// slices).  Symmetrized bf16 W_self slice resident in LDS (132 KB of the
// WGP's 320 KB) -> weights leave HBM exactly once.  h is exchanged between
// WGs as bf16 through a ping-pong buffer in d_ws; staging is a raw 32 KB/WG
// copy done with CDNA5 async global->LDS (ASYNCcnt) when available.  Per-step
// sync: one 16-WG atomic phase barrier per batch-tile (tiles are independent).
// ---------------------------------------------------------------------------
__global__ __launch_bounds__(128, 1) void scan_kernel(const float* __restrict__ Wl,
                                                      float* out,
                                                      __bf16* hexch,     // 2 x [32][1024]
                                                      unsigned* cnt) {   // 2 counters
  constexpr int N = 1024, T = 512, LDW = 1032;  // pad LDS rows across banks
  constexpr int HEXN = 32 * 1024;               // elements per ping-pong buffer
  __shared__ __bf16 Wlds[64 * LDW];             // Wlds[j][k] = Wsym[k][n0+j]
  __shared__ __bf16 hlds[16 * LDW];             // hlds[row][k] = h[b0+row][k]

  const int mt = blockIdx.x >> 4;               // batch tile: 0,1
  const int n0 = (blockIdx.x & 15) << 6;        // column slice start

  // One-time: load + symmetrize W_self slice into LDS (bf16).
  for (int idx = threadIdx.x; idx < 64 * 1024; idx += 128) {
    const int j = idx >> 10, k = idx & 1023;
    const int gj = n0 + j;
    Wlds[j * LDW + k] =
        (__bf16)(0.5f * (Wl[(size_t)k * N + gj] + Wl[(size_t)gj * N + k]));
  }

  const int wave = threadIdx.x >> 5;
  const int lane = threadIdx.x & 31;
  const int jloc = wave << 4;
  const int col = lane & 15;
  const int mb = (lane >> 4) << 3;
  const size_t bstride = (size_t)T * N;
  float* const obase0 = out + (size_t)(mt * 16) * bstride + n0 + jloc;
  unsigned* const mycnt = cnt + mt * 64;        // separate cachelines per tile

  for (int t = 0; t < T; ++t) {
    __syncthreads();
    // Stage h_{t-1}: raw bf16 copy, 16 rows x 2 KB, global -> LDS.
    {
      const __bf16* hsrc = hexch + (size_t)((t + 1) & 1) * HEXN + (size_t)(mt * 16) * N;
#pragma unroll 4
      for (int q = threadIdx.x; q < 2048; q += 128) {   // 2048 x 16B
        const int row = q >> 7;
        const int c8 = (q & 127) << 3;
        const __bf16* src = hsrc + row * N + c8;
        __bf16* dst = &hlds[row * LDW + c8];
#if HAVE_ASYNC_LDS
        __builtin_amdgcn_global_load_async_to_lds_b128((v4i*)src, (lds_v4i*)dst, 0, 0);
#else
        *(uint4*)dst = *(const uint4*)src;
#endif
      }
#if HAVE_ASYNC_LDS
      wait_async_zero();
#endif
    }
    __syncthreads();

    float* const op = obase0 + (size_t)t * N;
    v8f c;
#pragma unroll
    for (int r = 0; r < 8; ++r)                 // C init = x_proj tile (in d_out)
      c[r] = op[(size_t)(mb + r) * bstride + col];

#pragma unroll 4
    for (int k0 = 0; k0 < N; k0 += 32) {
      const __bf16* pa = &hlds[(lane & 15) * LDW + k0 + ((lane >> 4) << 3)];
      const v8bf alo = *(const v8bf*)(pa);
      const v8bf ahi = *(const v8bf*)(pa + 16);
      const v16bf a = __builtin_shufflevector(alo, ahi,
          0,1,2,3,4,5,6,7,8,9,10,11,12,13,14,15);
      const __bf16* pb = &Wlds[(jloc + col) * LDW + k0 + ((lane >> 4) << 3)];
      const v8bf blo = *(const v8bf*)(pb);
      const v8bf bhi = *(const v8bf*)(pb + 16);
      const v16bf b = __builtin_shufflevector(blo, bhi,
          0,1,2,3,4,5,6,7,8,9,10,11,12,13,14,15);
      c = WMMA_BF16(a, b, c);
    }

    // h_t = tanh(.) -> f32 output + bf16 exchange buffer buf[t&1].
    __bf16* const hw = hexch + (size_t)(t & 1) * HEXN;
#pragma unroll
    for (int r = 0; r < 8; ++r) {
      const float v = TANHF(c[r]);
      op[(size_t)(mb + r) * bstride + col] = v;
      hw[(size_t)(mt * 16 + mb + r) * N + n0 + jloc + col] = (__bf16)v;
    }

    // Per-tile step barrier (release stores, arrive, spin, acquire).
    __threadfence();
    __syncthreads();
    if (threadIdx.x == 0) {
      __hip_atomic_fetch_add(mycnt, 1u, __ATOMIC_RELEASE, __HIP_MEMORY_SCOPE_AGENT);
      const unsigned tgt = 16u * (unsigned)(t + 1);
      while (__hip_atomic_load(mycnt, __ATOMIC_ACQUIRE, __HIP_MEMORY_SCOPE_AGENT) < tgt)
        __builtin_amdgcn_s_sleep(1);
    }
    __syncthreads();
    __threadfence();   // acquire: drop WGP$ lines before next h-stage
  }
}

// Zero the barrier counters and seed ping-pong buffer 1 with bf16(h0).
__global__ void init_kernel(const float* __restrict__ h0, __bf16* hexch, unsigned* cnt) {
  const int i = blockIdx.x * blockDim.x + threadIdx.x;
  if (i == 0) { cnt[0] = 0u; cnt[64] = 0u; }
  if (i < 32 * 1024) hexch[32 * 1024 + i] = (__bf16)h0[i];
}

extern "C" void kernel_launch(void* const* d_in, const int* in_sizes, int n_in,
                              void* d_out, int out_size, void* d_ws, size_t ws_size,
                              hipStream_t stream) {
  (void)in_sizes; (void)n_in; (void)out_size; (void)ws_size;
  const float* X  = (const float*)d_in[0];  // input_sequence [32,512,1024]
  const float* H0 = (const float*)d_in[1];  // initial_state  [32,1024]
  const float* Wl = (const float*)d_in[2];  // W_self_lower   [1024,1024]
  const float* Wi = (const float*)d_in[3];  // W_input        [1024,1024]
  float* out = (float*)d_out;               // [32,512,1024] f32

  // d_ws layout: [0,512)B: two barrier counters (separate cachelines);
  //              [512, 512 + 128KB): bf16 h ping-pong exchange (2 x 32 x 1024).
  unsigned* cnt = (unsigned*)d_ws;
  __bf16* hexch = (__bf16*)((char*)d_ws + 512);

  init_kernel<<<128, 256, 0, stream>>>(H0, hexch, cnt);
  xproj_kernel<<<2048, 256, 0, stream>>>(X, Wi, out);
  scan_kernel<<<32, 128, 0, stream>>>(Wl, out, hexch, cnt);
}
